// RoPESelfAttention_22823456211211
// MI455X (gfx1250) — compile-verified, matrix operand order checked by reading, not run on hardware
//
#include <hip/hip_runtime.h>
#include <math.h>

// ---------------- types ----------------
typedef __bf16 bf16;
typedef __bf16   v16bf __attribute__((ext_vector_type(16)));
typedef __bf16   v8bf  __attribute__((ext_vector_type(8)));
typedef float    v8f   __attribute__((ext_vector_type(8)));
typedef unsigned v8u   __attribute__((ext_vector_type(8)));

#define B_  2
#define T_  2048
#define D_  1024
#define H_  16
#define DH_ 64

static __device__ __forceinline__ v8f zero8() {
  v8f z;
#pragma unroll
  for (int i = 0; i < 8; ++i) z[i] = 0.0f;
  return z;
}

static __device__ __forceinline__ v16bf make_frag(v8bf a, v8bf b) {
  v16bf r;
#pragma unroll
  for (int j = 0; j < 8; ++j) { r[j] = a[j]; r[8 + j] = b[j]; }
  return r;
}

static __device__ __forceinline__ v8f wmma_bf16(v16bf a, v16bf b, v8f c) {
  return __builtin_amdgcn_wmma_f32_16x16x32_bf16(false, a, false, b, (short)0, c,
                                                 false, false);
}

static __device__ __forceinline__ void split2(float x, bf16& h, bf16& l) {
  h = (bf16)x;
  l = (bf16)(x - (float)h);
}

// ---------------- prep kernels ----------------
__global__ __launch_bounds__(256) void split_bf16_k(const float* __restrict__ in,
                                                    bf16* __restrict__ h,
                                                    bf16* __restrict__ l, int n) {
  int i = blockIdx.x * blockDim.x + threadIdx.x;
  if (i < n) {
    float x = in[i];
    bf16 hh = (bf16)x;
    h[i] = hh;
    l[i] = (bf16)(x - (float)hh);
  }
}

// W is [K,N] row-major; write hi/lo transposed [N,K]
__global__ __launch_bounds__(256) void transpose_split_k(const float* __restrict__ W,
                                                         bf16* __restrict__ hT,
                                                         bf16* __restrict__ lT,
                                                         int K, int N) {
  int i = blockIdx.x * blockDim.x + threadIdx.x;
  if (i < N * K) {
    int n = i / K, k = i % K;
    float x = W[(size_t)k * N + n];
    bf16 hh = (bf16)x;
    hT[i] = hh;
    lT[i] = (bf16)(x - (float)hh);
  }
}

// ---------------- kernel 1: QKV GEMM + fused RoPE ----------------
// wave tile 32x64 (2 M-tiles x 4 N-tiles); WG = 8 waves stacked in M -> 256x64
// Issue order per k-step: HH pass, HL pass, LH pass over all 8 accumulators
// (same-accumulator reuse distance 8 -> covers the WMMA->WMMA hazard window).
__global__ __launch_bounds__(256) void qkv_rope_gemm(
    const bf16* __restrict__ xh, const bf16* __restrict__ xl,
    const bf16* __restrict__ wth, const bf16* __restrict__ wtl,
    bf16* __restrict__ Qh, bf16* __restrict__ Ql,
    bf16* __restrict__ Kh, bf16* __restrict__ Kl,
    bf16* __restrict__ Vth, bf16* __restrict__ Vtl) {
  const int tid = threadIdx.x, wave = tid >> 5, lane = tid & 31;
  const int l16 = lane & 15, lh = lane >> 4;
  const int Nbase = blockIdx.x * 64;
  const int Mbase = blockIdx.y * 256 + wave * 32;
  const int K = D_;

  const bf16* arow_h[2];
  const bf16* arow_l[2];
#pragma unroll
  for (int mt = 0; mt < 2; ++mt) {
    arow_h[mt] = xh + (size_t)(Mbase + mt * 16 + l16) * K;
    arow_l[mt] = xl + (size_t)(Mbase + mt * 16 + l16) * K;
  }
  const bf16* brow_h[4];
  const bf16* brow_l[4];
#pragma unroll
  for (int nt = 0; nt < 4; ++nt) {
    brow_h[nt] = wth + (size_t)(Nbase + nt * 16 + l16) * K;
    brow_l[nt] = wtl + (size_t)(Nbase + nt * 16 + l16) * K;
  }

  v8f acc[2][4];
#pragma unroll
  for (int mt = 0; mt < 2; ++mt)
#pragma unroll
    for (int nt = 0; nt < 4; ++nt) acc[mt][nt] = zero8();

  for (int kk = 0; kk < K; kk += 32) {
    const int o0 = kk + 8 * lh, o1 = kk + 16 + 8 * lh;
    v16bf aH[2], aL[2];
#pragma unroll
    for (int mt = 0; mt < 2; ++mt) {
      aH[mt] = make_frag(*(const v8bf*)(arow_h[mt] + o0),
                         *(const v8bf*)(arow_h[mt] + o1));
      aL[mt] = make_frag(*(const v8bf*)(arow_l[mt] + o0),
                         *(const v8bf*)(arow_l[mt] + o1));
    }
    v16bf bH[4], bL[4];
#pragma unroll
    for (int nt = 0; nt < 4; ++nt) {
      bH[nt] = make_frag(*(const v8bf*)(brow_h[nt] + o0),
                         *(const v8bf*)(brow_h[nt] + o1));
      bL[nt] = make_frag(*(const v8bf*)(brow_l[nt] + o0),
                         *(const v8bf*)(brow_l[nt] + o1));
    }
#pragma unroll
    for (int nt = 0; nt < 4; ++nt)
#pragma unroll
      for (int mt = 0; mt < 2; ++mt)
        acc[mt][nt] = wmma_bf16(aH[mt], bH[nt], acc[mt][nt]);
#pragma unroll
    for (int nt = 0; nt < 4; ++nt)
#pragma unroll
      for (int mt = 0; mt < 2; ++mt)
        acc[mt][nt] = wmma_bf16(aH[mt], bL[nt], acc[mt][nt]);
#pragma unroll
    for (int nt = 0; nt < 4; ++nt)
#pragma unroll
      for (int mt = 0; mt < 2; ++mt)
        acc[mt][nt] = wmma_bf16(aL[mt], bH[nt], acc[mt][nt]);
  }

  // ---- epilogue ----
  const int sect = Nbase >> 10;          // 0=Q 1=K 2=V
  const int hIdx = (Nbase & 1023) >> 6;  // head index
  const int b = Mbase >> 11;             // batch (32-row tile never crosses)

#pragma unroll
  for (int mt = 0; mt < 2; ++mt) {
    const int tBase = (Mbase & 2047) + mt * 16 + 8 * lh;
    if (sect == 2) {
      // V: store transposed [B,H,Dh,T] bf16 hi/lo; t contiguous over r.
#pragma unroll
      for (int nt = 0; nt < 4; ++nt) {
        int dh = nt * 16 + l16;
        size_t base = (((size_t)(b * H_ + hIdx)) * DH_ + dh) * T_ + tBase;
        v8bf ph, pl;
#pragma unroll
        for (int r = 0; r < 8; ++r) {
          bf16 hh, ll;
          split2(acc[mt][nt][r], hh, ll);
          ph[r] = hh;
          pl[r] = ll;
        }
        *(v8bf*)(Vth + base) = ph;
        *(v8bf*)(Vtl + base) = pl;
      }
    } else {
      bf16* dH = sect ? Kh : Qh;
      bf16* dL = sect ? Kl : Ql;
#pragma unroll
      for (int nt = 0; nt < 2; ++nt) {
        int i = nt * 16 + l16;  // rotary pair index 0..31
        float theta = (float)exp((double)i * -0.28782313662425575);  // 10000^(-i/32)
#pragma unroll
        for (int r = 0; r < 8; ++r) {
          int t = tBase + r;
          float s, c;
          sincosf((float)t * theta, &s, &c);
          float q1 = acc[mt][nt][r], q2 = acc[mt][nt + 2][r];
          float o1 = q1 * c + q2 * s;
          float o2 = q2 * c - q1 * s;
          size_t rowbase = (((size_t)(b * H_ + hIdx)) * T_ + t) * DH_;
          bf16 h1, l1, h2, l2;
          split2(o1, h1, l1);
          split2(o2, h2, l2);
          dH[rowbase + i] = h1;
          dL[rowbase + i] = l1;
          dH[rowbase + i + 32] = h2;
          dL[rowbase + i + 32] = l2;
        }
      }
    }
  }
}

// ---------------- kernel 2: attention (16 query rows / WG) ----------------
// S lives in LDS as raw 32-bit slots: fp32 logits -> exp -> packed {bf16 hi | bf16 lo}
__global__ __launch_bounds__(256) void attn_kernel(
    const bf16* __restrict__ Qh, const bf16* __restrict__ Ql,
    const bf16* __restrict__ Kh, const bf16* __restrict__ Kl,
    const bf16* __restrict__ Vth, const bf16* __restrict__ Vtl,
    bf16* __restrict__ AOh, bf16* __restrict__ AOl) {
  __shared__ unsigned S[16][T_];   // 128 KB (CDNA5 WGP: 320 KB LDS)
  __shared__ float Red[4][8][32];  // cross-wave K-split reduction

  const int tid = threadIdx.x, wave = tid >> 5, lane = tid & 31;
  const int l16 = lane & 15, lh = lane >> 4;
  const int q0 = blockIdx.x * 16;
  const int h = blockIdx.y, b = blockIdx.z;
  const int kLen = q0 + 16;
  const int kPad = (kLen + 31) & ~31;

  const size_t headQK = ((size_t)(b * H_ + h)) * T_ * DH_;
  const size_t headV = ((size_t)(b * H_ + h)) * DH_ * T_;

  // Q fragments for this 16-row block (K-dim = Dh = 64 -> 2 k-steps)
  const bf16* qrh = Qh + headQK + (size_t)(q0 + l16) * DH_;
  const bf16* qrl = Ql + headQK + (size_t)(q0 + l16) * DH_;
  v16bf aH[2], aL[2];
#pragma unroll
  for (int ks = 0; ks < 2; ++ks) {
    int o0 = ks * 32 + 8 * lh, o1 = ks * 32 + 16 + 8 * lh;
    aH[ks] = make_frag(*(const v8bf*)(qrh + o0), *(const v8bf*)(qrh + o1));
    aL[ks] = make_frag(*(const v8bf*)(qrl + o0), *(const v8bf*)(qrl + o1));
  }

  // ---- S = Q K^T * scale, causal-masked; only tiles with t_k <= q0+15 ----
  const float scale = 0.125f;  // 1/sqrt(64)
  for (int nt = wave; nt * 16 < kLen; nt += 8) {
    const bf16* krh = Kh + headQK + (size_t)(nt * 16 + l16) * DH_;
    const bf16* krl = Kl + headQK + (size_t)(nt * 16 + l16) * DH_;
    // three independent accumulator chains (covers WMMA->WMMA latency)
    v8f aHH = zero8(), aHL = zero8(), aLH = zero8();
#pragma unroll
    for (int ks = 0; ks < 2; ++ks) {
      int o0 = ks * 32 + 8 * lh, o1 = ks * 32 + 16 + 8 * lh;
      v16bf bH = make_frag(*(const v8bf*)(krh + o0), *(const v8bf*)(krh + o1));
      v16bf bL = make_frag(*(const v8bf*)(krl + o0), *(const v8bf*)(krl + o1));
      aHH = wmma_bf16(aH[ks], bH, aHH);
      aHL = wmma_bf16(aH[ks], bL, aHL);
      aLH = wmma_bf16(aL[ks], bH, aLH);
    }
    v8f acc = aHH + aHL + aLH;
    const int tk = nt * 16 + l16;
#pragma unroll
    for (int r = 0; r < 8; ++r) {
      int M = r + 8 * lh;
      float v = acc[r] * scale;
      if (tk > q0 + M) v = -1e9f;
      S[M][tk] = __builtin_bit_cast(unsigned, v);
    }
  }
  if (kPad > kLen) {  // pad to K=32 multiple with -1e9 (-> P = 0)
    int row = tid >> 4, c = kLen + (tid & 15);
    S[row][c] = __builtin_bit_cast(unsigned, -1e9f);
  }
  __syncthreads();

  // ---- softmax: 16 lanes per row, wave32 shfl reductions ----
  // final pass converts P once per element and packs {hi|lo} bf16 in place
  {
    int row = tid >> 4, sub = tid & 15;
    float m = -3.4e38f;
    for (int c = sub; c < kPad; c += 16)
      m = fmaxf(m, __builtin_bit_cast(float, S[row][c]));
#pragma unroll
    for (int off = 8; off >= 1; off >>= 1) m = fmaxf(m, __shfl_xor(m, off, 32));
    float sum = 0.0f;
    for (int c = sub; c < kPad; c += 16) {
      float e = __expf(__builtin_bit_cast(float, S[row][c]) - m);
      S[row][c] = __builtin_bit_cast(unsigned, e);
      sum += e;
    }
#pragma unroll
    for (int off = 8; off >= 1; off >>= 1) sum += __shfl_xor(sum, off, 32);
    float inv = 1.0f / sum;
    for (int c = sub; c < kPad; c += 16) {
      float p = __builtin_bit_cast(float, S[row][c]) * inv;
      bf16 hh, ll;
      split2(p, hh, ll);
      unsigned pk = ((unsigned)__builtin_bit_cast(unsigned short, ll) << 16) |
                    (unsigned)__builtin_bit_cast(unsigned short, hh);
      S[row][c] = pk;
    }
  }
  __syncthreads();

  // ---- out = P @ V : wave -> (d-tile = wave&3, K-half = wave>>2) ----
  const int nt = wave & 3, kh = wave >> 2;
  const int dcol = nt * 16 + l16;
  const bf16* vrh = Vth + headV + (size_t)dcol * T_;
  const bf16* vrl = Vtl + headV + (size_t)dcol * T_;
  v8f aHH = zero8(), aHL = zero8(), aLH = zero8();
  for (int ci = kh; ci * 32 < kPad; ci += 2) {
    int kk = ci * 32;
    int o0 = kk + 8 * lh, o1 = kk + 16 + 8 * lh;
    // A from LDS: 2x ds_load_b128 of packed {hi|lo}, even/odd extract
    v16bf i0 = __builtin_bit_cast(v16bf, *(const v8u*)&S[l16][o0]);
    v16bf i1 = __builtin_bit_cast(v16bf, *(const v8u*)&S[l16][o1]);
    v8bf h0 = __builtin_shufflevector(i0, i0, 0, 2, 4, 6, 8, 10, 12, 14);
    v8bf l0 = __builtin_shufflevector(i0, i0, 1, 3, 5, 7, 9, 11, 13, 15);
    v8bf h1 = __builtin_shufflevector(i1, i1, 0, 2, 4, 6, 8, 10, 12, 14);
    v8bf l1 = __builtin_shufflevector(i1, i1, 1, 3, 5, 7, 9, 11, 13, 15);
    v16bf pH = make_frag(h0, h1);
    v16bf pL = make_frag(l0, l1);
    v16bf bH = make_frag(*(const v8bf*)(vrh + o0), *(const v8bf*)(vrh + o1));
    v16bf bL = make_frag(*(const v8bf*)(vrl + o0), *(const v8bf*)(vrl + o1));
    aHH = wmma_bf16(pH, bH, aHH);
    aHL = wmma_bf16(pH, bL, aHL);
    aLH = wmma_bf16(pL, bH, aLH);
  }
  v8f acc = aHH + aHL + aLH;
  if (kh == 1) {
#pragma unroll
    for (int r = 0; r < 8; ++r) Red[nt][r][lane] = acc[r];
  }
  __syncthreads();
  if (kh == 0) {
#pragma unroll
    for (int r = 0; r < 8; ++r) acc[r] += Red[nt][r][lane];
    int col = h * DH_ + dcol;
#pragma unroll
    for (int r = 0; r < 8; ++r) {
      int row = b * T_ + q0 + r + 8 * lh;
      bf16 hh, ll;
      split2(acc[r], hh, ll);
      AOh[(size_t)row * D_ + col] = hh;
      AOl[(size_t)row * D_ + col] = ll;
    }
  }
}

// ---------------- kernel 3: projection GEMM -> fp32 d_out ----------------
// wave tile 32x64, WG 256x64; same 3-pass issue order as kernel 1
__global__ __launch_bounds__(256) void proj_gemm(
    const bf16* __restrict__ ah, const bf16* __restrict__ al,
    const bf16* __restrict__ wth, const bf16* __restrict__ wtl,
    float* __restrict__ out) {
  const int tid = threadIdx.x, wave = tid >> 5, lane = tid & 31;
  const int l16 = lane & 15, lh = lane >> 4;
  const int Nbase = blockIdx.x * 64;
  const int Mbase = blockIdx.y * 256 + wave * 32;
  const int K = D_;

  const bf16* arow_h[2];
  const bf16* arow_l[2];
#pragma unroll
  for (int mt = 0; mt < 2; ++mt) {
    arow_h[mt] = ah + (size_t)(Mbase + mt * 16 + l16) * K;
    arow_l[mt] = al + (size_t)(Mbase + mt * 16 + l16) * K;
  }
  const bf16* brow_h[4];
  const bf16* brow_l[4];
#pragma unroll
  for (int nt = 0; nt < 4; ++nt) {
    brow_h[nt] = wth + (size_t)(Nbase + nt * 16 + l16) * K;
    brow_l[nt] = wtl + (size_t)(Nbase + nt * 16 + l16) * K;
  }

  v8f acc[2][4];
#pragma unroll
  for (int mt = 0; mt < 2; ++mt)
#pragma unroll
    for (int nt = 0; nt < 4; ++nt) acc[mt][nt] = zero8();

  for (int kk = 0; kk < K; kk += 32) {
    const int o0 = kk + 8 * lh, o1 = kk + 16 + 8 * lh;
    v16bf aH[2], aL[2];
#pragma unroll
    for (int mt = 0; mt < 2; ++mt) {
      aH[mt] = make_frag(*(const v8bf*)(arow_h[mt] + o0),
                         *(const v8bf*)(arow_h[mt] + o1));
      aL[mt] = make_frag(*(const v8bf*)(arow_l[mt] + o0),
                         *(const v8bf*)(arow_l[mt] + o1));
    }
    v16bf bH[4], bL[4];
#pragma unroll
    for (int nt = 0; nt < 4; ++nt) {
      bH[nt] = make_frag(*(const v8bf*)(brow_h[nt] + o0),
                         *(const v8bf*)(brow_h[nt] + o1));
      bL[nt] = make_frag(*(const v8bf*)(brow_l[nt] + o0),
                         *(const v8bf*)(brow_l[nt] + o1));
    }
#pragma unroll
    for (int nt = 0; nt < 4; ++nt)
#pragma unroll
      for (int mt = 0; mt < 2; ++mt)
        acc[mt][nt] = wmma_bf16(aH[mt], bH[nt], acc[mt][nt]);
#pragma unroll
    for (int nt = 0; nt < 4; ++nt)
#pragma unroll
      for (int mt = 0; mt < 2; ++mt)
        acc[mt][nt] = wmma_bf16(aH[mt], bL[nt], acc[mt][nt]);
#pragma unroll
    for (int nt = 0; nt < 4; ++nt)
#pragma unroll
      for (int mt = 0; mt < 2; ++mt)
        acc[mt][nt] = wmma_bf16(aL[mt], bH[nt], acc[mt][nt]);
  }

#pragma unroll
  for (int mt = 0; mt < 2; ++mt)
#pragma unroll
    for (int nt = 0; nt < 4; ++nt) {
      int col = Nbase + nt * 16 + l16;
#pragma unroll
      for (int r = 0; r < 8; ++r) {
        size_t row = (size_t)(Mbase + mt * 16 + 8 * lh + r);
        out[row * D_ + col] = acc[mt][nt][r];
      }
    }
}

// ---------------- host launcher ----------------
extern "C" void kernel_launch(void* const* d_in, const int* in_sizes, int n_in,
                              void* d_out, int out_size, void* d_ws, size_t ws_size,
                              hipStream_t stream) {
  (void)in_sizes; (void)n_in; (void)out_size; (void)ws_size;
  const float* x = (const float*)d_in[0];
  // d_in[1] = mask (triu ones) -> applied analytically as causal mask
  const float* W_qkv = (const float*)d_in[2];
  const float* W_proj = (const float*)d_in[3];
  float* out = (float*)d_out;

  const size_t ME = (size_t)B_ * T_ * D_;        // 4194304
  const size_t WQ = (size_t)D_ * 3 * D_;         // 3145728
  const size_t WP = (size_t)D_ * D_;             // 1048576
  const size_t HE = (size_t)B_ * H_ * T_ * DH_;  // 4194304

  char* p = (char*)d_ws;
  auto take = [&](size_t bytes) { char* r = p; p += (bytes + 255) & ~(size_t)255; return r; };
  bf16* xh = (bf16*)take(ME * 2);
  bf16* xl = (bf16*)take(ME * 2);
  bf16* wqh = (bf16*)take(WQ * 2);
  bf16* wql = (bf16*)take(WQ * 2);
  bf16* wph = (bf16*)take(WP * 2);
  bf16* wpl = (bf16*)take(WP * 2);
  bf16* Qh = (bf16*)take(HE * 2);
  bf16* Ql = (bf16*)take(HE * 2);
  bf16* Kh = (bf16*)take(HE * 2);
  bf16* Kl = (bf16*)take(HE * 2);
  bf16* Vth = (bf16*)take(HE * 2);
  bf16* Vtl = (bf16*)take(HE * 2);
  bf16* AOh = (bf16*)take(ME * 2);
  bf16* AOl = (bf16*)take(ME * 2);

  split_bf16_k<<<(int)((ME + 255) / 256), 256, 0, stream>>>(x, xh, xl, (int)ME);
  transpose_split_k<<<(int)((WQ + 255) / 256), 256, 0, stream>>>(W_qkv, wqh, wql, D_, 3 * D_);
  transpose_split_k<<<(int)((WP + 255) / 256), 256, 0, stream>>>(W_proj, wph, wpl, D_, D_);

  qkv_rope_gemm<<<dim3(48, 16), 256, 0, stream>>>(xh, xl, wqh, wql, Qh, Ql, Kh, Kl,
                                                  Vth, Vtl);

  attn_kernel<<<dim3(T_ / 16, H_, B_), 256, 0, stream>>>(Qh, Ql, Kh, Kl, Vth, Vtl,
                                                         AOh, AOl);

  proj_gemm<<<dim3(16, 16), 256, 0, stream>>>(AOh, AOl, wph, wpl, out);
}
